// MLA_35287451304411
// MI455X (gfx1250) — matrix operand order repaired and checked by reading, not compiled, not run on hardware
//
#include <hip/hip_runtime.h>
#include <math.h>

typedef unsigned short u16;
typedef __attribute__((ext_vector_type(16))) __bf16 v16bf;
typedef __attribute__((ext_vector_type(8)))  float  v8f;

#define B_    2
#define T_    1024
#define C_    2048
#define H_    16
#define NOPE_ 128
#define ROPE_ 64
#define VD_   128
#define QKD_  192
#define QLR_  1536
#define KVLR_ 512
#define M_    (B_*T_)
#define LORA_SCALE_ 2.0f
#define SCALE_ 0.07216878364870323f
#define EPS_   1e-6f

// ---------- helpers ----------
__device__ __forceinline__ u16 f2bf(float f){
  unsigned u = __float_as_uint(f);
  u += 0x7fffu + ((u >> 16) & 1u);
  return (u16)(u >> 16);
}

union frag_u { uint4 q[2]; v16bf v; };

// A fragment (16x32 bf16, M x K): lane row = lane&15, hi = lane>>4 selects K-half.
// element e: k = 16*hi + e  -> two contiguous 16B chunks.
__device__ __forceinline__ v16bf load_a_frag(const u16* rowp, int k0, int hi){
  frag_u r;
  r.q[0] = *(const uint4*)(rowp + k0 + 16*hi);
  r.q[1] = *(const uint4*)(rowp + k0 + 16*hi + 8);
  return r.v;
}
// B fragment (32x16 bf16, K x N) from N-major (transposed) storage:
// lane col = lane&15, element e: k = 16*hi + e -> one contiguous 32B chunk.
__device__ __forceinline__ v16bf load_b_frag(const u16* rowp, int k0, int hi){
  const uint4* p = (const uint4*)(rowp + k0 + 16*hi);
  frag_u r; r.q[0] = p[0]; r.q[1] = p[1];
  return r.v;
}

__device__ __forceinline__ v8f wmma_bf16(v16bf a, v16bf b, v8f c){
  return __builtin_amdgcn_wmma_f32_16x16x32_bf16(false, a, false, b, (short)0, c,
                                                 false, false);
}

// ---------- elementwise kernels ----------
__global__ __launch_bounds__(256)
void k_convert_bf16(const float* __restrict__ in, u16* __restrict__ out, size_t n){
  for (size_t i = (size_t)blockIdx.x*256 + threadIdx.x; i < n;
       i += (size_t)gridDim.x*256)
    out[i] = f2bf(in[i]);
}

// in: f32 (K,N) row-major -> out: bf16 (N,Kpad), zero pad for k in [K,Kpad)
__global__ __launch_bounds__(256)
void k_transpose_bf16(const float* __restrict__ in, u16* __restrict__ out,
                      int K, int N, int Kpad){
  size_t total = (size_t)N * Kpad;
  for (size_t i = (size_t)blockIdx.x*256 + threadIdx.x; i < total;
       i += (size_t)gridDim.x*256){
    int n = (int)(i / Kpad), k = (int)(i % Kpad);
    out[i] = (k < K) ? f2bf(in[(size_t)k*N + n]) : (u16)0;
  }
}

// f32 (M,Kin) -> bf16 (M,Kpad), zero pad
__global__ __launch_bounds__(256)
void k_pad_bf16(const float* __restrict__ in, u16* __restrict__ out,
                int M, int Kin, int Kpad){
  size_t total = (size_t)M * Kpad;
  for (size_t i = (size_t)blockIdx.x*256 + threadIdx.x; i < total;
       i += (size_t)gridDim.x*256){
    int m = (int)(i / Kpad), k = (int)(i % Kpad);
    out[i] = (k < Kin) ? f2bf(in[(size_t)m*Kin + k]) : (u16)0;
  }
}

// rmsnorm row-wise: in f32 (row stride ld_in, n cols) -> out bf16 (stride n)
__global__ __launch_bounds__(256)
void k_rmsnorm(const float* __restrict__ in, int ld_in,
               const float* __restrict__ w, u16* __restrict__ out, int n){
  const int row = blockIdx.x;
  const float* x = in + (size_t)row*ld_in;
  __shared__ float red[8];
  float s = 0.f;
  for (int c = threadIdx.x; c < n; c += 256){ float v = x[c]; s += v*v; }
  s += __shfl_xor(s, 1, 32);  s += __shfl_xor(s, 2, 32);
  s += __shfl_xor(s, 4, 32);  s += __shfl_xor(s, 8, 32);
  s += __shfl_xor(s, 16, 32);
  if ((threadIdx.x & 31) == 0) red[threadIdx.x >> 5] = s;
  __syncthreads();
  if (threadIdx.x == 0){
    float tot = 0.f;
#pragma unroll
    for (int i = 0; i < 8; i++) tot += red[i];
    red[0] = rsqrtf(tot / (float)n + EPS_);
  }
  __syncthreads();
  const float r = red[0];
  for (int c = threadIdx.x; c < n; c += 256)
    out[(size_t)row*n + c] = f2bf(w[c]*x[c]*r);
}

// ---------- generic bf16 WMMA GEMM: C = alpha*(A @ BT^T) [+ bias] [+ C] ----------
// A: bf16 (M,K) row-major, lda. BT: bf16 (N,K) row-major (i.e. B transposed), ldb.
// C: f32 (M,N), ldc. Block = 256 thr = 8 waves (4x2), block tile 128x128,
// wave tile 32x64 (2x4 of 16x16). Row/col clamping makes any M,N legal.
__global__ __launch_bounds__(256)
void k_gemm_bf16(const u16* __restrict__ A, int lda,
                 const u16* __restrict__ BT, int ldb,
                 float* __restrict__ C, int ldc,
                 const float* __restrict__ bias,
                 int M, int N, int K, float alpha, int accum){
  const int wave = threadIdx.x >> 5, lane = threadIdx.x & 31;
  const int hi = lane >> 4, ln = lane & 15;
  const int wm = wave >> 1, wn = wave & 1;
  const int row0 = blockIdx.y*128 + wm*32;
  const int col0 = blockIdx.x*128 + wn*64;

  const u16* arow[2];
#pragma unroll
  for (int i = 0; i < 2; i++){
    int r = row0 + 16*i + ln; if (r > M-1) r = M-1;
    arow[i] = A + (size_t)r*lda;
  }
  const u16* brow[4];
#pragma unroll
  for (int j = 0; j < 4; j++){
    int c = col0 + 16*j + ln; if (c > N-1) c = N-1;
    brow[j] = BT + (size_t)c*ldb;
  }

  v8f acc[2][4];
#pragma unroll
  for (int i = 0; i < 2; i++)
#pragma unroll
    for (int j = 0; j < 4; j++)
#pragma unroll
      for (int r = 0; r < 8; r++) acc[i][j][r] = 0.0f;

  for (int k0 = 0; k0 < K; k0 += 32){
    v16bf af[2], bf_[4];
#pragma unroll
    for (int i = 0; i < 2; i++) af[i]  = load_a_frag(arow[i], k0, hi);
#pragma unroll
    for (int j = 0; j < 4; j++) bf_[j] = load_b_frag(brow[j], k0, hi);
#pragma unroll
    for (int i = 0; i < 2; i++)
#pragma unroll
      for (int j = 0; j < 4; j++)
        acc[i][j] = wmma_bf16(af[i], bf_[j], acc[i][j]);
  }

#pragma unroll
  for (int i = 0; i < 2; i++)
#pragma unroll
    for (int j = 0; j < 4; j++){
      const int n = col0 + 16*j + ln;
      if (n >= N) continue;
      const float bv = bias ? bias[n] : 0.0f;
#pragma unroll
      for (int r = 0; r < 8; r++){
        const int m = row0 + 16*i + r + 8*hi;
        if (m < M){
          size_t idx = (size_t)m*ldc + n;
          float v = alpha*acc[i][j][r] + bv;
          if (accum) v += C[idx];
          C[idx] = v;
        }
      }
    }
}

// ---------- assembly kernels (LoRA add + RoPE + layout change to bf16) ----------
// Q: qfull f32 (M, H*192) -> Qbf (B,H,T,192) bf16
__global__ __launch_bounds__(256)
void k_assemble_q(const float* __restrict__ qfull, const float* __restrict__ xq,
                  const float* __restrict__ q_lb, const float* __restrict__ fcos,
                  const float* __restrict__ fsin, u16* __restrict__ Qout){
  const int m = blockIdx.x, b = m >> 10, t = m & (T_-1);
  const float* qr  = qfull + (size_t)m*(H_*QKD_);
  const float* xqr = xq + (size_t)m*16;
  for (int col = threadIdx.x; col < H_*QKD_; col += 256){
    const int hh = col / QKD_, d = col % QKD_;
    float v;
    if (d < NOPE_){
      float acc = 0.f;
#pragma unroll
      for (int j = 0; j < 16; j++) acc += xqr[j]*q_lb[j*(H_*NOPE_) + hh*NOPE_ + d];
      v = qr[col] + LORA_SCALE_*acc;
    } else {
      const int dd = d - NOPE_, pr = dd >> 1;
      const float c = fcos[t*(ROPE_/2) + pr], s = fsin[t*(ROPE_/2) + pr];
      const float xr = qr[hh*QKD_ + NOPE_ + pr*2];
      const float xi = qr[hh*QKD_ + NOPE_ + pr*2 + 1];
      v = (dd & 1) ? (xr*s + xi*c) : (xr*c - xi*s);
    }
    Qout[((size_t)(b*H_ + hh)*T_ + t)*QKD_ + d] = f2bf(v);
  }
}

// K: kv2 f32 (M, H*256) nope part + RoPE(kv[:,512:576]) broadcast -> Kbf (B,H,T,192)
__global__ __launch_bounds__(256)
void k_assemble_k(const float* __restrict__ kv2, const float* __restrict__ kvraw,
                  const float* __restrict__ xk, const float* __restrict__ k_lb,
                  const float* __restrict__ fcos, const float* __restrict__ fsin,
                  u16* __restrict__ Kout){
  const int m = blockIdx.x, b = m >> 10, t = m & (T_-1);
  const float* kr  = kv2 + (size_t)m*(H_*(NOPE_+VD_));
  const float* pe  = kvraw + (size_t)m*(KVLR_+ROPE_) + KVLR_;
  const float* xkr = xk + (size_t)m*16;
  for (int col = threadIdx.x; col < H_*QKD_; col += 256){
    const int hh = col / QKD_, d = col % QKD_;
    float v;
    if (d < NOPE_){
      float acc = 0.f;
#pragma unroll
      for (int j = 0; j < 16; j++) acc += xkr[j]*k_lb[j*(H_*NOPE_) + hh*NOPE_ + d];
      v = kr[hh*(NOPE_+VD_) + d] + LORA_SCALE_*acc;
    } else {
      const int dd = d - NOPE_, pr = dd >> 1;
      const float c = fcos[t*(ROPE_/2) + pr], s = fsin[t*(ROPE_/2) + pr];
      const float xr = pe[pr*2], xi = pe[pr*2 + 1];
      v = (dd & 1) ? (xr*s + xi*c) : (xr*c - xi*s);
    }
    Kout[((size_t)(b*H_ + hh)*T_ + t)*QKD_ + d] = f2bf(v);
  }
}

// V: kv2 f32 (M, H*256) v part + LoRA -> Vt bf16 (B,H,VD,T)  (transposed for WMMA B)
__global__ __launch_bounds__(256)
void k_assemble_v(const float* __restrict__ kv2, const float* __restrict__ xv,
                  const float* __restrict__ v_lb, u16* __restrict__ Vt){
  const int m = blockIdx.x, b = m >> 10, t = m & (T_-1);
  const float* kr  = kv2 + (size_t)m*(H_*(NOPE_+VD_));
  const float* xvr = xv + (size_t)m*16;
  for (int col = threadIdx.x; col < H_*VD_; col += 256){
    const int hh = col >> 7, d = col & 127;
    float acc = 0.f;
#pragma unroll
    for (int j = 0; j < 16; j++) acc += xvr[j]*v_lb[j*(H_*VD_) + col];
    const float v = kr[hh*(NOPE_+VD_) + NOPE_ + d] + LORA_SCALE_*acc;
    Vt[((size_t)(b*H_ + hh)*VD_ + d)*T_ + t] = f2bf(v);
  }
}

// ---------- flash attention (causal), WMMA, one wave per 16 query rows ----------
__global__ __launch_bounds__(256)
void k_attn(const u16* __restrict__ Qbf, const u16* __restrict__ Kbf,
            const u16* __restrict__ Vt, u16* __restrict__ ybf){
  __shared__ __align__(16) u16 pstage[8][16][32];
  const int wave = threadIdx.x >> 5, lane = threadIdx.x & 31;
  const int hi = lane >> 4, ln = lane & 15;
  const int b = blockIdx.z, h = blockIdx.y;
  const int q0 = blockIdx.x*128 + wave*16;

  const u16* Qrow  = Qbf + ((size_t)(b*H_ + h)*T_ + q0 + ln)*QKD_;
  const u16* Kbase = Kbf + (size_t)(b*H_ + h)*T_*QKD_;
  const u16* Vbase = Vt  + (size_t)(b*H_ + h)*VD_*T_;

  v16bf qf[6];
#pragma unroll
  for (int i = 0; i < 6; i++) qf[i] = load_a_frag(Qrow, i*32, hi);

  v8f Oacc[8];
#pragma unroll
  for (int j = 0; j < 8; j++)
#pragma unroll
    for (int r = 0; r < 8; r++) Oacc[j][r] = 0.0f;

  float mrow[8], lrow[8];
#pragma unroll
  for (int r = 0; r < 8; r++){ mrow[r] = -INFINITY; lrow[r] = 0.0f; }

  const int ntiles = (q0 + 15)/32 + 1;
  for (int tile = 0; tile < ntiles; ++tile){
    const int kcol = tile*32;
    v8f s0, s1;
#pragma unroll
    for (int r = 0; r < 8; r++){ s0[r] = 0.f; s1[r] = 0.f; }
    const u16* Kr0 = Kbase + (size_t)(kcol + ln)*QKD_;
    const u16* Kr1 = Kr0 + 16*QKD_;
#pragma unroll
    for (int i = 0; i < 6; i++){
      v16bf b0 = load_b_frag(Kr0, i*32, hi);
      v16bf b1 = load_b_frag(Kr1, i*32, hi);
      s0 = wmma_bf16(qf[i], b0, s0);
      s1 = wmma_bf16(qf[i], b1, s1);
    }
    float p0[8], p1[8], rsv[8];
#pragma unroll
    for (int r = 0; r < 8; r++){
      const int m = q0 + r + 8*hi;
      float a0 = (kcol + ln      <= m) ? s0[r]*SCALE_ : -INFINITY;
      float a1 = (kcol + 16 + ln <= m) ? s1[r]*SCALE_ : -INFINITY;
      float mx = fmaxf(a0, a1);
      mx = fmaxf(mx, __shfl_xor(mx, 1, 32));
      mx = fmaxf(mx, __shfl_xor(mx, 2, 32));
      mx = fmaxf(mx, __shfl_xor(mx, 4, 32));
      mx = fmaxf(mx, __shfl_xor(mx, 8, 32));
      mx = fmaxf(mx, mrow[r]);
      float e0 = 0.f, e1 = 0.f, rs = 0.f;
      if (mx != -INFINITY){
        e0 = (a0 == -INFINITY) ? 0.f : __expf(a0 - mx);
        e1 = (a1 == -INFINITY) ? 0.f : __expf(a1 - mx);
        rs = (mrow[r] == -INFINITY) ? 0.f : __expf(mrow[r] - mx);
      }
      float rsum = e0 + e1;
      rsum += __shfl_xor(rsum, 1, 32);
      rsum += __shfl_xor(rsum, 2, 32);
      rsum += __shfl_xor(rsum, 4, 32);
      rsum += __shfl_xor(rsum, 8, 32);
      lrow[r] = lrow[r]*rs + rsum;
      mrow[r] = mx;
      p0[r] = e0; p1[r] = e1; rsv[r] = rs;
    }
#pragma unroll
    for (int j = 0; j < 8; j++)
#pragma unroll
      for (int r = 0; r < 8; r++) Oacc[j][r] *= rsv[r];

    // C-layout -> A-layout via per-wave LDS staging (same-wave DS ops are in-order)
#pragma unroll
    for (int r = 0; r < 8; r++){
      pstage[wave][r + 8*hi][ln]      = f2bf(p0[r]);
      pstage[wave][r + 8*hi][16 + ln] = f2bf(p1[r]);
    }
    v16bf pa = load_a_frag(&pstage[wave][ln][0], 0, hi);
#pragma unroll
    for (int j = 0; j < 8; j++){
      v16bf vb = load_b_frag(Vbase + (size_t)(j*16 + ln)*T_, kcol, hi);
      Oacc[j] = wmma_bf16(pa, vb, Oacc[j]);
    }
  }

#pragma unroll
  for (int r = 0; r < 8; r++){
    const int m = q0 + r + 8*hi;
    const float inv = 1.0f / lrow[r];
#pragma unroll
    for (int j = 0; j < 8; j++){
      const int col = h*VD_ + j*16 + ln;
      ybf[((size_t)b*T_ + m)*(size_t)C_ + col] = f2bf(Oacc[j][r]*inv);
    }
  }
}

// ---------- host ----------
extern "C" void kernel_launch(void* const* d_in, const int* in_sizes, int n_in,
                              void* d_out, int out_size, void* d_ws, size_t ws_size,
                              hipStream_t stream){
  const float* x       = (const float*)d_in[0];
  const float* fcos    = (const float*)d_in[1];
  const float* fsin    = (const float*)d_in[2];
  const float* wq_a    = (const float*)d_in[3];
  const float* bq_a    = (const float*)d_in[4];
  const float* q_nw    = (const float*)d_in[5];
  const float* wq_b    = (const float*)d_in[6];
  const float* bq_b    = (const float*)d_in[7];
  const float* wkv_a   = (const float*)d_in[8];
  const float* bkv_a   = (const float*)d_in[9];
  const float* kv_nw   = (const float*)d_in[10];
  const float* wkv_b   = (const float*)d_in[11];
  const float* bkv_b   = (const float*)d_in[12];
  const float* wo      = (const float*)d_in[13];
  const float* bo      = (const float*)d_in[14];
  const float* q_la    = (const float*)d_in[15];
  const float* q_lb    = (const float*)d_in[16];
  const float* k_la    = (const float*)d_in[17];
  const float* k_lb    = (const float*)d_in[18];
  const float* v_la    = (const float*)d_in[19];
  const float* v_lb    = (const float*)d_in[20];
  const float* o_la    = (const float*)d_in[21];
  const float* o_lb    = (const float*)d_in[22];
  float* out = (float*)d_out;

  char* ws = (char*)d_ws;
  size_t off = 0;
  auto alloc = [&](size_t bytes)->char*{
    off = (off + 255) & ~(size_t)255;
    char* p = ws + off; off += bytes; return p;
  };
  u16*   xbf   = (u16*)  alloc((size_t)M_*C_*2);
  u16*   wqaT  = (u16*)  alloc((size_t)QLR_*C_*2);
  u16*   wqbT  = (u16*)  alloc((size_t)(H_*QKD_)*QLR_*2);
  u16*   wkvaT = (u16*)  alloc((size_t)(KVLR_+ROPE_)*C_*2);
  u16*   wkvbT = (u16*)  alloc((size_t)(H_*(NOPE_+VD_))*KVLR_*2);
  u16*   woT   = (u16*)  alloc((size_t)C_*C_*2);
  u16*   qlaT  = (u16*)  alloc((size_t)16*C_*2);
  u16*   klaT  = (u16*)  alloc((size_t)16*C_*2);
  u16*   vlaT  = (u16*)  alloc((size_t)16*C_*2);
  u16*   olaT  = (u16*)  alloc((size_t)16*C_*2);
  u16*   olbT  = (u16*)  alloc((size_t)C_*32*2);       // (2048, 32) K-padded
  float* qa    = (float*)alloc((size_t)M_*QLR_*4);
  u16*   qn    = (u16*)  alloc((size_t)M_*QLR_*2);
  float* qfull = (float*)alloc((size_t)M_*(H_*QKD_)*4);
  float* kvb   = (float*)alloc((size_t)M_*(KVLR_+ROPE_)*4);
  u16*   kvn   = (u16*)  alloc((size_t)M_*KVLR_*2);
  float* kv2   = (float*)alloc((size_t)M_*(H_*(NOPE_+VD_))*4);
  float* xq    = (float*)alloc((size_t)M_*16*4);
  float* xk    = (float*)alloc((size_t)M_*16*4);
  float* xv    = (float*)alloc((size_t)M_*16*4);
  float* yo    = (float*)alloc((size_t)M_*16*4);
  u16*   yobf  = (u16*)  alloc((size_t)M_*32*2);
  u16*   Qbf   = (u16*)  alloc((size_t)B_*H_*T_*QKD_*2);
  u16*   Kbf   = (u16*)  alloc((size_t)B_*H_*T_*QKD_*2);
  u16*   Vt    = (u16*)  alloc((size_t)B_*H_*VD_*T_*2);
  u16*   ybf   = (u16*)  alloc((size_t)M_*C_*2);
  (void)ws_size; (void)in_sizes; (void)n_in; (void)out_size;

  auto gsz = [](size_t n){ return (unsigned)((n + 255)/256); };

  // 1. convert activations + transpose/convert weights to bf16 N-major
  k_convert_bf16<<<gsz((size_t)M_*C_), 256, 0, stream>>>(x, xbf, (size_t)M_*C_);
  k_transpose_bf16<<<gsz((size_t)QLR_*C_), 256, 0, stream>>>(wq_a,  wqaT,  C_, QLR_, C_);
  k_transpose_bf16<<<gsz((size_t)(H_*QKD_)*QLR_), 256, 0, stream>>>(wq_b, wqbT, QLR_, H_*QKD_, QLR_);
  k_transpose_bf16<<<gsz((size_t)(KVLR_+ROPE_)*C_), 256, 0, stream>>>(wkv_a, wkvaT, C_, KVLR_+ROPE_, C_);
  k_transpose_bf16<<<gsz((size_t)(H_*(NOPE_+VD_))*KVLR_), 256, 0, stream>>>(wkv_b, wkvbT, KVLR_, H_*(NOPE_+VD_), KVLR_);
  k_transpose_bf16<<<gsz((size_t)C_*C_), 256, 0, stream>>>(wo, woT, C_, C_, C_);
  k_transpose_bf16<<<gsz((size_t)16*C_), 256, 0, stream>>>(q_la, qlaT, C_, 16, C_);
  k_transpose_bf16<<<gsz((size_t)16*C_), 256, 0, stream>>>(k_la, klaT, C_, 16, C_);
  k_transpose_bf16<<<gsz((size_t)16*C_), 256, 0, stream>>>(v_la, vlaT, C_, 16, C_);
  k_transpose_bf16<<<gsz((size_t)16*C_), 256, 0, stream>>>(o_la, olaT, C_, 16, C_);
  k_transpose_bf16<<<gsz((size_t)C_*32), 256, 0, stream>>>(o_lb, olbT, 16, C_, 32);

  auto gemm = [&](const u16* A, int lda, const u16* BT, int ldb, float* Cp, int ldc,
                  const float* bias, int M, int N, int K, float alpha, int accum){
    dim3 grid((N+127)/128, (M+127)/128);
    k_gemm_bf16<<<grid, 256, 0, stream>>>(A, lda, BT, ldb, Cp, ldc, bias, M, N, K, alpha, accum);
  };

  // 2. projections
  gemm(xbf, C_, wqaT,  C_, qa,  QLR_,          bq_a,  M_, QLR_,          C_, 1.0f, 0);
  gemm(xbf, C_, wkvaT, C_, kvb, KVLR_+ROPE_,   bkv_a, M_, KVLR_+ROPE_,   C_, 1.0f, 0);
  gemm(xbf, C_, qlaT,  C_, xq,  16,            nullptr, M_, 16,          C_, 1.0f, 0);
  gemm(xbf, C_, klaT,  C_, xk,  16,            nullptr, M_, 16,          C_, 1.0f, 0);
  gemm(xbf, C_, vlaT,  C_, xv,  16,            nullptr, M_, 16,          C_, 1.0f, 0);

  // 3. rmsnorm
  k_rmsnorm<<<M_, 256, 0, stream>>>(qa,  QLR_,        q_nw,  qn,  QLR_);
  k_rmsnorm<<<M_, 256, 0, stream>>>(kvb, KVLR_+ROPE_, kv_nw, kvn, KVLR_);

  // 4. second-stage projections
  gemm(qn,  QLR_,  wqbT,  QLR_,  qfull, H_*QKD_,        bq_b,  M_, H_*QKD_,        QLR_,  1.0f, 0);
  gemm(kvn, KVLR_, wkvbT, KVLR_, kv2,   H_*(NOPE_+VD_), bkv_b, M_, H_*(NOPE_+VD_), KVLR_, 1.0f, 0);

  // 5. assemble Q/K/V in attention layouts (LoRA add + RoPE fused)
  k_assemble_q<<<M_, 256, 0, stream>>>(qfull, xq, q_lb, fcos, fsin, Qbf);
  k_assemble_k<<<M_, 256, 0, stream>>>(kv2, kvb, xk, k_lb, fcos, fsin, Kbf);
  k_assemble_v<<<M_, 256, 0, stream>>>(kv2, xv, v_lb, Vt);

  // 6. causal flash attention
  k_attn<<<dim3(T_/128, H_, B_), 256, 0, stream>>>(Qbf, Kbf, Vt, ybf);

  // 7. output projection + o-LoRA
  gemm(ybf, C_, olaT, C_, yo, 16, nullptr, M_, 16, C_, 1.0f, 0);
  k_pad_bf16<<<gsz((size_t)M_*32), 256, 0, stream>>>(yo, yobf, M_, 16, 32);
  gemm(ybf,  C_, woT,  C_, out, C_, bo,      M_, C_, C_, 1.0f, 0);
  gemm(yobf, 32, olbT, 32, out, C_, nullptr, M_, C_, 32, LORA_SCALE_, 1);
}
